// GNN_34411277976463
// MI455X (gfx1250) — compile-verified, matrix operand order checked by reading, not run on hardware
//
#include <hip/hip_runtime.h>
#include <hip/hip_bf16.h>

// ---------------- vector types (plain ext vectors: safe in unions) ----------
typedef __attribute__((ext_vector_type(16))) __bf16       v16bf;
typedef __attribute__((ext_vector_type(8)))  float        v8f;
typedef __attribute__((ext_vector_type(4)))  unsigned int u32x4;
typedef __attribute__((ext_vector_type(4)))  float        f32x4;
typedef __attribute__((ext_vector_type(2)))  float        f32x2;

union BFrag { v16bf v; u32x4 q[2]; };

__device__ __forceinline__ unsigned short f2bf(float f) {
    unsigned int u = __float_as_uint(f);
    u += 0x7FFFu + ((u >> 16) & 1u);          // round-to-nearest-even
    return (unsigned short)(u >> 16);
}

// ---------------- small utility kernels -------------------------------------
__global__ void k_fill_f32(float* __restrict__ p, int n, float v) {
    int i = blockIdx.x * blockDim.x + threadIdx.x;
    if (i < n) p[i] = v;
}

__global__ void k_degree(const int* __restrict__ dst, float* __restrict__ deg, int nE) {
    int i = blockIdx.x * blockDim.x + threadIdx.x;
    if (i < nE) unsafeAtomicAdd(&deg[dst[i]], 1.0f);
}

__global__ void k_rsqrt(const float* __restrict__ deg, float* __restrict__ dinv, int n) {
    int i = blockIdx.x * blockDim.x + threadIdx.x;
    if (i < n) dinv[i] = rsqrtf(deg[i]);       // deg >= 1 (self-loop), no zero case
}

__global__ void k_convert_x(const float* __restrict__ in, unsigned short* __restrict__ out, int n) {
    int i = blockIdx.x * blockDim.x + threadIdx.x;
    if (i < n) out[i] = f2bf(in[i]);
}

// out_bf16 = bf16(relu(in + bias[c]))  (feeds next layer's GEMM A matrix)
__global__ void k_bias_relu_bf16(const float* __restrict__ in, const float* __restrict__ bias,
                                 unsigned short* __restrict__ out, int n, int cmask) {
    int i = blockIdx.x * blockDim.x + threadIdx.x;
    if (i < n) {
        float v = in[i] + bias[i & cmask];
        out[i] = f2bf(v > 0.0f ? v : 0.0f);
    }
}

// W: [128 x C] row-major f32  ->  Wt: [C x 128] bf16 (transposed for contiguous B frags)
__global__ void k_convert_wt(const float* __restrict__ W, unsigned short* __restrict__ Wt, int C) {
    int i = blockIdx.x * blockDim.x + threadIdx.x;
    int total = 128 * C;
    if (i < total) {
        int k = i / C, c = i % C;
        Wt[c * 128 + k] = f2bf(W[i]);
    }
}

__global__ void k_bias_add(float* __restrict__ out, const float* __restrict__ bias, int n, int cmask) {
    int i = blockIdx.x * blockDim.x + threadIdx.x;
    if (i < n) out[i] += bias[i & cmask];
}

// ---------------- WMMA GEMM: H[N x COUT] = Abf[N x 128] * Wt^T --------------
// Register-persistent weights: each wave owns a (16*NTW)-column group of W and
// keeps all 4*NTW B fragments (128 VGPRs at NTW=4) live across a grid-stride
// loop over 16-row tiles. Per tile: b128 clause loads of the A fragments,
// then 4 WMMAs issue back-to-back per wait; next tile's A row is prefetched.
//
// __launch_bounds__(256, 4): 4 waves/SIMD -> 256-VGPR budget so the ~210-VGPR
// working set (128 B + 32 A + 32 acc + addressing) stays spill-free.
//
// 16-bit A 16x32 lane layout (ISA 7.12.2): lane l<16 -> row m0+l, K kb..kb+7
// and kb+16..kb+23 with kb = k0 (+8 for lanes 16..31). B mirrors per column,
// served contiguously from the pre-transposed Wt.
template<int COUT, int NTW>
__global__ __launch_bounds__(256, 4)
void k_gemm(const unsigned short* __restrict__ Abf,
            const unsigned short* __restrict__ Wt,
            float* __restrict__ H, int nRows) {
    const int lane    = threadIdx.x & 31;
    const int wpb     = blockDim.x >> 5;
    const int gwave   = blockIdx.x * wpb + (threadIdx.x >> 5);
    const int n0g     = blockIdx.y * (NTW * 16);          // column group base
    const int halfSel = lane >> 4;
    const int l16     = lane & 15;
    const int nTiles  = (nRows + 15) >> 4;
    const int stride  = gridDim.x * wpb;

    // ---- pin B fragments in registers: b[k-step][column-tile] ----
    BFrag b[4][NTW];
#pragma unroll
    for (int k = 0; k < 4; ++k) {
        const int kb = k * 32 + halfSel * 8;
#pragma unroll
        for (int nt = 0; nt < NTW; ++nt) {
            const unsigned short* wCol = Wt + (size_t)(n0g + nt * 16 + l16) * 128 + kb;
            b[k][nt].q[0] = *(const u32x4*)(wCol);
            b[k][nt].q[1] = *(const u32x4*)(wCol + 16);
        }
    }

    for (int tile = gwave; tile < nTiles; tile += stride) {   // wave-uniform loop
        const int m0 = tile << 4;
        int rowA = m0 + l16;
        if (rowA >= nRows) rowA = nRows - 1;                  // stores are guarded
        const unsigned short* aRow = Abf + (size_t)rowA * 128;

        // prefetch next tile's A row (2 x 128B lines per 256B row)
        const int ntile = tile + stride;
        if (ntile < nTiles) {
            const unsigned short* pref = Abf + (size_t)(ntile * 16 + l16) * 128;
            __builtin_prefetch(pref, 0, 3);
            __builtin_prefetch(pref + 64, 0, 3);
        }

        BFrag a[4];
#pragma unroll
        for (int k = 0; k < 4; ++k) {
            const int kb = k * 32 + halfSel * 8;
            a[k].q[0] = *(const u32x4*)(aRow + kb);
            a[k].q[1] = *(const u32x4*)(aRow + kb + 16);
        }

        v8f acc[NTW];
#pragma unroll
        for (int nt = 0; nt < NTW; ++nt)
#pragma unroll
            for (int j = 0; j < 8; ++j) acc[nt][j] = 0.0f;

#pragma unroll
        for (int k = 0; k < 4; ++k)
#pragma unroll
            for (int nt = 0; nt < NTW; ++nt)
                acc[nt] = __builtin_amdgcn_wmma_f32_16x16x32_bf16(
                    false, a[k].v, false, b[k][nt].v, (short)0, acc[nt], false, false);

        // C/D layout: VGPR r -> row m0+r (lanes 0-15) / m0+8+r (lanes 16-31).
        // Wave-uniform full-tile test: fast path stores unguarded (no per-store
        // EXEC juggling) and keeps EXEC all-1s for the next iteration's WMMAs.
        const int mBase = m0 + halfSel * 8;
        float* hOut = H + (size_t)mBase * COUT + n0g + l16;
        if (m0 + 16 <= nRows) {
#pragma unroll
            for (int nt = 0; nt < NTW; ++nt)
#pragma unroll
                for (int r = 0; r < 8; ++r)
                    hOut[(size_t)r * COUT + nt * 16] = acc[nt][r];
        } else {
#pragma unroll
            for (int nt = 0; nt < NTW; ++nt)
#pragma unroll
                for (int r = 0; r < 8; ++r)
                    if (mBase + r < nRows)
                        hOut[(size_t)r * COUT + nt * 16] = acc[nt][r];
        }
    }
}

// ---------------- edge aggregation: Agg[dst] += H[src] * norm ---------------
// One wave per edge (incl. implicit self-loops at e >= nE). Lanes cover the C
// channels: coalesced 512B/256B row gather from L2-resident H, then native
// f32 global atomics into the (also L2-resident) accumulator.
template<int C>
__global__ void k_aggregate(const float* __restrict__ H,
                            const int* __restrict__ srcA,
                            const int* __restrict__ dstA,
                            const float* __restrict__ dinv,
                            float* __restrict__ Agg,
                            int nE, int nN) {
    const int lane = threadIdx.x & 31;
    const int e    = blockIdx.x * (blockDim.x >> 5) + (threadIdx.x >> 5);
    const int total = nE + nN;
    if (e >= total) return;
    int s, d;
    if (e < nE) { s = srcA[e]; d = dstA[e]; }
    else        { s = d = e - nE; }             // self-loop
    const float nrm = dinv[s] * dinv[d];
    constexpr int PER = C / 32;                  // 4 (C=128) or 2 (C=64)
    const float* hp = H   + (size_t)s * C + lane * PER;
    float*       ap = Agg + (size_t)d * C + lane * PER;
    if constexpr (PER == 4) {
        const f32x4 v = *(const f32x4*)hp;
        unsafeAtomicAdd(ap + 0, v.x * nrm);
        unsafeAtomicAdd(ap + 1, v.y * nrm);
        unsafeAtomicAdd(ap + 2, v.z * nrm);
        unsafeAtomicAdd(ap + 3, v.w * nrm);
    } else {
        const f32x2 v = *(const f32x2*)hp;
        unsafeAtomicAdd(ap + 0, v.x * nrm);
        unsafeAtomicAdd(ap + 1, v.y * nrm);
    }
}

// ---------------- driver ----------------------------------------------------
extern "C" void kernel_launch(void* const* d_in, const int* in_sizes, int n_in,
                              void* d_out, int out_size, void* d_ws, size_t ws_size,
                              hipStream_t stream) {
    const float* x  = (const float*)d_in[0];
    const int*   ei = (const int*)  d_in[1];
    const float* W0 = (const float*)d_in[2];
    const float* b0 = (const float*)d_in[3];
    const float* W1 = (const float*)d_in[4];
    const float* b1 = (const float*)d_in[5];
    const float* W2 = (const float*)d_in[6];
    const float* b2 = (const float*)d_in[7];
    float* out = (float*)d_out;

    const int N = in_sizes[0] / 128;
    const int E = in_sizes[1] / 2;
    const int* srcA = ei;
    const int* dstA = ei + E;

    // workspace carve-out (~130 MB total)
    char* base = (char*)d_ws;
    size_t off = 0;
    auto carve = [&](size_t bytes) -> void* {
        void* p = base + off;
        off += (bytes + 255) & ~(size_t)255;
        return p;
    };
    float*          deg  = (float*)carve((size_t)N * 4);
    float*          dinv = (float*)carve((size_t)N * 4);
    unsigned short* Abf  = (unsigned short*)carve((size_t)N * 128 * 2);
    float*          H    = (float*)carve((size_t)N * 128 * 4);
    float*          Agg  = (float*)carve((size_t)N * 128 * 4);
    unsigned short* Wt0  = (unsigned short*)carve(128 * 128 * 2);
    unsigned short* Wt1  = (unsigned short*)carve(128 * 128 * 2);
    unsigned short* Wt2  = (unsigned short*)carve(128 * 64 * 2);
    (void)ws_size; (void)n_in; (void)out_size;

    const int T = 256;
    const int totalEdges = E + N;
    const int aggBlocks  = (totalEdges + 7) / 8;   // 8 waves/block, 1 edge/wave

    // GEMM grid: x = row-tile waves (grid-stride, ~3 tiles/wave so the pinned
    // B registers amortize), y = column groups of 64.
    const dim3 gemmGrid128(256, 2);   // 2048 waves/group over 6250 tiles
    const dim3 gemmGrid64 (256, 1);

    // --- normalization: deg = 1 (self-loop) + scatter count; dinv = rsqrt ---
    k_fill_f32<<<(N + T - 1) / T, T, 0, stream>>>(deg, N, 1.0f);
    k_degree  <<<(E + T - 1) / T, T, 0, stream>>>(dstA, deg, E);
    k_rsqrt   <<<(N + T - 1) / T, T, 0, stream>>>(deg, dinv, N);

    // --- weight / input conversion to bf16 (Wt transposed) ------------------
    k_convert_wt<<<(128 * 128 + T - 1) / T, T, 0, stream>>>(W0, Wt0, 128);
    k_convert_wt<<<(128 * 128 + T - 1) / T, T, 0, stream>>>(W1, Wt1, 128);
    k_convert_wt<<<(128 * 64  + T - 1) / T, T, 0, stream>>>(W2, Wt2, 64);
    k_convert_x <<<((N * 128) + T - 1) / T, T, 0, stream>>>(x, Abf, N * 128);

    // --- layer 0: h = relu(scatter(norm * (x@W0)[src]) + b0) ----------------
    k_gemm<128, 4><<<gemmGrid128, 256, 0, stream>>>(Abf, Wt0, H, N);
    k_fill_f32 <<<((N * 128) + T - 1) / T, T, 0, stream>>>(Agg, N * 128, 0.0f);
    k_aggregate<128><<<aggBlocks, 256, 0, stream>>>(H, srcA, dstA, dinv, Agg, E, N);
    k_bias_relu_bf16<<<((N * 128) + T - 1) / T, T, 0, stream>>>(Agg, b0, Abf, N * 128, 127);

    // --- layer 1 ------------------------------------------------------------
    k_gemm<128, 4><<<gemmGrid128, 256, 0, stream>>>(Abf, Wt1, H, N);
    k_fill_f32 <<<((N * 128) + T - 1) / T, T, 0, stream>>>(Agg, N * 128, 0.0f);
    k_aggregate<128><<<aggBlocks, 256, 0, stream>>>(H, srcA, dstA, dinv, Agg, E, N);
    k_bias_relu_bf16<<<((N * 128) + T - 1) / T, T, 0, stream>>>(Agg, b1, Abf, N * 128, 127);

    // --- layer 2 (C_out = 64, no relu): aggregate straight into d_out -------
    k_gemm<64, 4><<<gemmGrid64, 256, 0, stream>>>(Abf, Wt2, H, N);
    k_fill_f32<<<((N * 64) + T - 1) / T, T, 0, stream>>>(out, N * 64, 0.0f);
    k_aggregate<64><<<aggBlocks, 256, 0, stream>>>(H, srcA, dstA, dinv, out, E, N);
    k_bias_add<<<((N * 64) + T - 1) / T, T, 0, stream>>>(out, b2, N * 64, 63);
}
